// GraphConvLayer_27728308863843
// MI455X (gfx1250) — matrix operand order repaired and checked by reading, not compiled, not run on hardware
//
#include <hip/hip_runtime.h>

typedef __attribute__((ext_vector_type(16))) _Float16 v16h;
typedef __attribute__((ext_vector_type(8)))  float    v8f;

#define N_NODES 50000
#define N_EDGES 800000
#define D_NODE  64
#define D_EDGE  32
#define D_HID   128

// ---------------- workspace layout (bytes) ----------------
// agg:   [N_NODES * D_NODE] f32
// then the four weight matrices pre-swizzled to WMMA B-tile layout (f16).
// One B tile (16x16x32 step) = 32 lanes * 16 halves * 2B = 1024 bytes.
#define AGG_FLOATS (N_NODES * D_NODE)          // 3,200,000
#define MW1_OFF  ((size_t)AGG_FLOATS * 4)      // 12,800,000 (1024-aligned)
#define MW1_TILES 40                            // K=160 -> 5 kt, N=128 -> 8 nt
#define MW2_OFF  (MW1_OFF + (size_t)MW1_TILES * 1024)
#define MW2_TILES 16                            // K=128 -> 4 kt, N=64 -> 4 nt
#define UW1_OFF  (MW2_OFF + (size_t)MW2_TILES * 1024)
#define UW1_TILES 32                            // K=128, N=128
#define UW2_OFF  (UW1_OFF + (size_t)UW1_TILES * 1024)
#define UW2_TILES 16                            // K=128, N=64
#define TOT_TILES (MW1_TILES + MW2_TILES + UW1_TILES + UW2_TILES)  // 104

// Contiguous K order (matches swizzled B): half h (0..15) within the v16h,
// hi = lane>>4 -> K = 16*hi + h. Returns K offset within the 32-wide K tile.
__device__ __forceinline__ constexpr int a_k(int h, int hi) {
    return 16 * hi + h;
}

// ---------------- kernel 1: zero aggregation buffer ----------------
__global__ void zero_agg(float* __restrict__ agg) {
    int i = blockIdx.x * 256 + threadIdx.x;
    if (i < AGG_FLOATS) agg[i] = 0.0f;
}

// ---------------- kernel 2: swizzle weights (f32 -> f16, B-tile layout) ---
// B 32x16 (KxN) f16 layout used here, contiguous K order: lanes 0-15 hold K=0..15 (halves 0..15),
// lanes 16-31 hold K=16..31; lane%16 = N column.
__global__ void swizzle_weights(const float* __restrict__ mW1,
                                const float* __restrict__ mW2,
                                const float* __restrict__ uW1,
                                const float* __restrict__ uW2,
                                _Float16* __restrict__ dst /* at MW1_OFF */) {
    int id = blockIdx.x * 256 + threadIdx.x;
    if (id >= TOT_TILES * 512) return;
    int tile = id >> 9;           // global tile index 0..103
    int r    = id & 511;
    int l    = r >> 4;            // lane 0..31
    int h    = r & 15;            // half 0..15
    const float* W; int Ntot, kt, nt;
    if (tile < 40)      { W = mW1; Ntot = 128; int t2 = tile;      kt = t2 >> 3; nt = t2 & 7; }
    else if (tile < 56) { W = mW2; Ntot = 64;  int t2 = tile - 40; kt = t2 >> 2; nt = t2 & 3; }
    else if (tile < 88) { W = uW1; Ntot = 128; int t2 = tile - 56; kt = t2 >> 3; nt = t2 & 7; }
    else                { W = uW2; Ntot = 64;  int t2 = tile - 88; kt = t2 >> 2; nt = t2 & 3; }
    int k = kt * 32 + 16 * (l >> 4) + h;
    int n = nt * 16 + (l & 15);
    dst[(size_t)tile * 512 + l * 16 + h] = (_Float16)W[k * Ntot + n];
}

// ---------------- kernel 3: per-edge message MLP + scatter-add ----------
// 4 waves / block, 1 wave = one 16-edge tile. LDS: 4*(2560+2048)*2 = 36,864B.
__global__ __launch_bounds__(128) void edge_msg_kernel(
    const float* __restrict__ nodes, const int* __restrict__ row,
    const int* __restrict__ col, const float* __restrict__ ef,
    const float* __restrict__ mb1, const float* __restrict__ mb2,
    const _Float16* __restrict__ w1sw, const _Float16* __restrict__ w2sw,
    float* __restrict__ agg) {
    __shared__ _Float16 sA[4][16 * 160];   // staged inputs  (16 edges x K=160)
    __shared__ _Float16 sH[4][16 * 128];   // hidden (16 x D_HID) post-ReLU
    const int lane = threadIdx.x & 31;
    const int wave = threadIdx.x >> 5;
    const int t    = blockIdx.x * 4 + wave;        // 50000 tiles exactly
    const int e0   = t * 16;
    _Float16* a  = sA[wave];
    _Float16* hh = sH[wave];

    // ---- stage [src(64) | dst(64) | edge(32)] rows, coalesced, f32->f16
#pragma unroll
    for (int e = 0; e < 16; ++e) {
        int rs = row[e0 + e];
        int cs = col[e0 + e];
        a[e * 160 + lane]       = (_Float16)nodes[rs * 64 + lane];
        a[e * 160 + 32 + lane]  = (_Float16)nodes[rs * 64 + 32 + lane];
        a[e * 160 + 64 + lane]  = (_Float16)nodes[cs * 64 + lane];
        a[e * 160 + 96 + lane]  = (_Float16)nodes[cs * 64 + 32 + lane];
        a[e * 160 + 128 + lane] = (_Float16)ef[(e0 + e) * 32 + lane];
    }

    const int m = lane & 15, hi = lane >> 4;

    // ---- A tiles for GEMM1 (K = 160 -> 5 tiles)
    v16h A[5];
#pragma unroll
    for (int kt = 0; kt < 5; ++kt)
#pragma unroll
        for (int h = 0; h < 16; ++h)
            A[kt][h] = a[m * 160 + kt * 32 + a_k(h, hi)];

    // ---- GEMM1: [16x160] x [160x128] + b1
    v8f acc[8];
#pragma unroll
    for (int nt = 0; nt < 8; ++nt) {
        float b = mb1[nt * 16 + m];
#pragma unroll
        for (int r = 0; r < 8; ++r) acc[nt][r] = b;
    }
#pragma unroll
    for (int kt = 0; kt < 5; ++kt) {
#pragma unroll
        for (int nt = 0; nt < 8; ++nt) {
            v16h B = *(const v16h*)(w1sw + (kt * 8 + nt) * 512 + lane * 16);
            acc[nt] = __builtin_amdgcn_wmma_f32_16x16x32_f16(
                false, A[kt], false, B, (short)0, acc[nt], false, false);
        }
    }

    // ---- ReLU, stage hidden as f16 (row-major 16x128)
#pragma unroll
    for (int nt = 0; nt < 8; ++nt)
#pragma unroll
        for (int r = 0; r < 8; ++r) {
            float v = acc[nt][r];
            hh[(r + 8 * hi) * 128 + nt * 16 + m] = (_Float16)(v > 0.0f ? v : 0.0f);
        }

    // ---- GEMM2: [16x128] x [128x64] + b2
    v8f acc2[4];
#pragma unroll
    for (int nt = 0; nt < 4; ++nt) {
        float b = mb2[nt * 16 + m];
#pragma unroll
        for (int r = 0; r < 8; ++r) acc2[nt][r] = b;
    }
#pragma unroll
    for (int kt = 0; kt < 4; ++kt) {
        v16h A2;
#pragma unroll
        for (int h = 0; h < 16; ++h)
            A2[h] = hh[m * 128 + kt * 32 + a_k(h, hi)];
#pragma unroll
        for (int nt = 0; nt < 4; ++nt) {
            v16h B = *(const v16h*)(w2sw + (kt * 4 + nt) * 512 + lane * 16);
            acc2[nt] = __builtin_amdgcn_wmma_f32_16x16x32_f16(
                false, A2, false, B, (short)0, acc2[nt], false, false);
        }
    }

    // ---- scatter-add messages to agg[col[e]]
#pragma unroll
    for (int r = 0; r < 8; ++r) {
        int cs = col[e0 + r + 8 * hi];
#pragma unroll
        for (int nt = 0; nt < 4; ++nt)
            __hip_atomic_fetch_add(&agg[cs * 64 + nt * 16 + m], acc2[nt][r],
                                   __ATOMIC_RELAXED, __HIP_MEMORY_SCOPE_AGENT);
    }
}

// ---------------- kernel 4: per-node update MLP ----------------
__global__ __launch_bounds__(128) void node_update_kernel(
    const float* __restrict__ nodes, const float* __restrict__ agg,
    const float* __restrict__ ub1, const float* __restrict__ ub2,
    const _Float16* __restrict__ w1sw, const _Float16* __restrict__ w2sw,
    float* __restrict__ out) {
    __shared__ _Float16 sA[4][16 * 128];
    __shared__ _Float16 sH[4][16 * 128];
    const int lane = threadIdx.x & 31;
    const int wave = threadIdx.x >> 5;
    int t = blockIdx.x * 4 + wave;
    if (t > 3124) t = 3124;                 // tail: duplicate tile, same stores
    const int n0 = t * 16;
    _Float16* a  = sA[wave];
    _Float16* hh = sH[wave];

#pragma unroll
    for (int e = 0; e < 16; ++e) {
        int nid = n0 + e;
        a[e * 128 + lane]      = (_Float16)nodes[nid * 64 + lane];
        a[e * 128 + 32 + lane] = (_Float16)nodes[nid * 64 + 32 + lane];
        a[e * 128 + 64 + lane] = (_Float16)agg[nid * 64 + lane];
        a[e * 128 + 96 + lane] = (_Float16)agg[nid * 64 + 32 + lane];
    }

    const int m = lane & 15, hi = lane >> 4;
    v16h A[4];
#pragma unroll
    for (int kt = 0; kt < 4; ++kt)
#pragma unroll
        for (int h = 0; h < 16; ++h)
            A[kt][h] = a[m * 128 + kt * 32 + a_k(h, hi)];

    v8f acc[8];
#pragma unroll
    for (int nt = 0; nt < 8; ++nt) {
        float b = ub1[nt * 16 + m];
#pragma unroll
        for (int r = 0; r < 8; ++r) acc[nt][r] = b;
    }
#pragma unroll
    for (int kt = 0; kt < 4; ++kt) {
#pragma unroll
        for (int nt = 0; nt < 8; ++nt) {
            v16h B = *(const v16h*)(w1sw + (kt * 8 + nt) * 512 + lane * 16);
            acc[nt] = __builtin_amdgcn_wmma_f32_16x16x32_f16(
                false, A[kt], false, B, (short)0, acc[nt], false, false);
        }
    }

#pragma unroll
    for (int nt = 0; nt < 8; ++nt)
#pragma unroll
        for (int r = 0; r < 8; ++r) {
            float v = acc[nt][r];
            hh[(r + 8 * hi) * 128 + nt * 16 + m] = (_Float16)(v > 0.0f ? v : 0.0f);
        }

    v8f acc2[4];
#pragma unroll
    for (int nt = 0; nt < 4; ++nt) {
        float b = ub2[nt * 16 + m];
#pragma unroll
        for (int r = 0; r < 8; ++r) acc2[nt][r] = b;
    }
#pragma unroll
    for (int kt = 0; kt < 4; ++kt) {
        v16h A2;
#pragma unroll
        for (int h = 0; h < 16; ++h)
            A2[h] = hh[m * 128 + kt * 32 + a_k(h, hi)];
#pragma unroll
        for (int nt = 0; nt < 4; ++nt) {
            v16h B = *(const v16h*)(w2sw + (kt * 4 + nt) * 512 + lane * 16);
            acc2[nt] = __builtin_amdgcn_wmma_f32_16x16x32_f16(
                false, A2, false, B, (short)0, acc2[nt], false, false);
        }
    }

#pragma unroll
    for (int r = 0; r < 8; ++r) {
        int nid = n0 + r + 8 * hi;
#pragma unroll
        for (int nt = 0; nt < 4; ++nt)
            out[nid * 64 + nt * 16 + m] = acc2[nt][r];
    }
}

extern "C" void kernel_launch(void* const* d_in, const int* in_sizes, int n_in,
                              void* d_out, int out_size, void* d_ws, size_t ws_size,
                              hipStream_t stream) {
    const float* nodes = (const float*)d_in[0];
    const int*   row   = (const int*)d_in[1];
    const int*   col   = (const int*)d_in[2];
    const float* ef    = (const float*)d_in[3];
    const float* mW1   = (const float*)d_in[4];
    const float* mb1   = (const float*)d_in[5];
    const float* mW2   = (const float*)d_in[6];
    const float* mb2   = (const float*)d_in[7];
    const float* uW1   = (const float*)d_in[8];
    const float* ub1   = (const float*)d_in[9];
    const float* uW2   = (const float*)d_in[10];
    const float* ub2   = (const float*)d_in[11];
    float* out = (float*)d_out;
    char*  ws  = (char*)d_ws;
    float* agg = (float*)(ws + 0);

    zero_agg<<<(AGG_FLOATS + 255) / 256, 256, 0, stream>>>(agg);
    swizzle_weights<<<(TOT_TILES * 512 + 255) / 256, 256, 0, stream>>>(
        mW1, mW2, uW1, uW2, (_Float16*)(ws + MW1_OFF));
    edge_msg_kernel<<<N_EDGES / (16 * 4), 128, 0, stream>>>(
        nodes, row, col, ef, mb1, mb2,
        (const _Float16*)(ws + MW1_OFF), (const _Float16*)(ws + MW2_OFF), agg);
    node_update_kernel<<<(3125 + 3) / 4, 128, 0, stream>>>(
        nodes, agg, ub1, ub2,
        (const _Float16*)(ws + UW1_OFF), (const _Float16*)(ws + UW2_OFF), out);
}
